// NeRFNetwork_30107720745471
// MI455X (gfx1250) — compile-verified, matrix operand order checked by reading, not run on hardware
//
#include <hip/hip_runtime.h>
#include <math.h>

typedef __attribute__((ext_vector_type(16))) _Float16 v16h;
typedef __attribute__((ext_vector_type(8)))  float    v8f;

#define NUM_LEVELS   16
#define TABLE_SIZE   (1u << 19)
#define T_SAMPLES    192
#define PTS_PER_BLOCK 256

// SCALES[l] = 2 * 2^(2l/3) - 1 ; RES[l] = ceil(SCALES[l]) + 1
constexpr float kScales[NUM_LEVELS] = {
    1.0f,           2.17480210f,   4.03968420f,   7.0f,
    11.69920842f,   19.15873680f,  31.0f,         49.79683366f,
    79.63494719f,   127.0f,        202.18733465f, 321.53978877f,
    511.0f,         811.74933861f, 1289.15915509f, 2047.0f};
constexpr int kRes[NUM_LEVELS] = {
    2, 4, 6, 8, 13, 21, 32, 51, 81, 128, 204, 323, 512, 813, 1291, 2048};

// softplus(10x)/10, branch-free stable form:
// softplus(y) = max(y,0) + log(1 + exp(-|y|))
__device__ __forceinline__ float softplus10(float x) {
    float y = 10.0f * x;
    float e = __expf(-fabsf(y));          // v_exp_f32
    float sp = fmaxf(y, 0.0f) + __logf(1.0f + e);  // v_log_f32
    return sp * 0.1f;
}

// K index inside a 16-half (v16h) fragment for 16-bit WMMA operands.
// i = half index 0..15, g = lane/16.
__device__ __forceinline__ int kmap(int i, int g) {
    int v = i >> 1, b = i & 1;
    int base = (v < 4) ? (2 * v + b) : (16 + 2 * (v - 4) + b);
    return base + 8 * g;
}

union HF { v16h v; _Float16 e[16]; };

__global__ __launch_bounds__(PTS_PER_BLOCK)
void nerf_density_fused(const float* __restrict__ x,
                        const float* __restrict__ emb,
                        const float* __restrict__ W0,
                        const float* __restrict__ b0,
                        const float* __restrict__ W1,
                        const float* __restrict__ b1,
                        float* __restrict__ out, int P) {
    __shared__ _Float16 sFeats[PTS_PER_BLOCK * 64];      // 32 KB
    // W0 pre-swizzled into WMMA B-fragment order:
    // fragment f = kt*4+nt (kt: K-half, nt: N-tile); per lane 16 contiguous halves
    __shared__ __align__(16) _Float16 sW0f[8 * 32 * 16]; // 8 KB

    const int tid = threadIdx.x;

    // ---- Stage W0 into LDS, swizzled to fragment layout ----
    {
        const int f  = tid >> 5;   // fragment 0..7
        const int l  = tid & 31;   // lane within fragment
        const int kt = f >> 2;     // K-half (0: k<32, 1: k>=32)
        const int nt = f & 3;      // N tile
        const int fg = l >> 4;
        const int fn = l & 15;
        _Float16* dst = &sW0f[(f * 32 + l) * 16];
        #pragma unroll
        for (int i = 0; i < 16; ++i) {
            const int k = kt * 32 + kmap(i, fg);
            dst[i] = (_Float16)W0[k * 64 + nt * 16 + fn];
        }
    }

    // ---- Hash-grid encode: one point per thread ----
    const int p = blockIdx.x * PTS_PER_BLOCK + tid;
    if (p < P) {
        const int ray = p / T_SAMPLES;
        const int t   = p - ray * T_SAMPLES;
        const float* xr = x + (size_t)ray * T_SAMPLES * 2;
        const float ox = xr[0], oy = xr[1];
        float dx = xr[(T_SAMPLES - 1) * 2 + 0] - ox;
        float dy = xr[(T_SAMPLES - 1) * 2 + 1] - oy;
        const float inv = 1.0f / sqrtf(dx * dx + dy * dy);
        dx *= inv; dy *= inv;
        const float z = 2.0f * (float)t / (float)(T_SAMPLES - 1);
        const float X = fminf(fmaxf(ox + dx * z, -1.0f), 1.0f);
        const float Y = fminf(fmaxf(oy + dy * z, -1.0f), 1.0f);
        const float ux = (X + 1.0f) * 0.5f;
        const float uy = (Y + 1.0f) * 0.5f;

        _Float16* fout = &sFeats[tid * 64];
        #pragma unroll
        for (int l = 0; l < NUM_LEVELS; ++l) {
            const float s   = kScales[l];
            const int   res = kRes[l];
            const bool hashed = ((long long)res * res > (long long)TABLE_SIZE);
            const float px = ux * s + 0.5f, py = uy * s + 0.5f;
            const float fx = floorf(px),   fy = floorf(py);
            const float frx = px - fx,     fry = py - fy;
            const int ix = (int)fx, iy = (int)fy;
            const float* tbl = emb + (size_t)l * TABLE_SIZE * 4u;
            float a0 = 0.f, a1 = 0.f, a2 = 0.f, a3 = 0.f;
            #pragma unroll
            for (int c = 0; c < 4; ++c) {
                const int ddx = c & 1, ddy = c >> 1;
                const int cx = ix + ddx, cy = iy + ddy;
                unsigned idx;
                if (hashed)
                    idx = (((unsigned)cx) ^ (((unsigned)cy) * 2654435761u)) &
                          (TABLE_SIZE - 1u);
                else
                    idx = (unsigned)(cy * res + cx);
                const float w = (ddx ? frx : 1.0f - frx) *
                                (ddy ? fry : 1.0f - fry);
                const float4 v =
                    *reinterpret_cast<const float4*>(tbl + (size_t)idx * 4u);
                a0 += w * v.x; a1 += w * v.y; a2 += w * v.z; a3 += w * v.w;
            }
            fout[l * 4 + 0] = (_Float16)a0;
            fout[l * 4 + 1] = (_Float16)a1;
            fout[l * 4 + 2] = (_Float16)a2;
            fout[l * 4 + 3] = (_Float16)a3;
        }
    }
    __syncthreads();

    // ---- MLP layer 0 via v_wmma_f32_16x16x32_f16, fused epilogue ----
    const int wave = tid >> 5;
    const int lane = tid & 31;
    const int g    = lane >> 4;
    const int n0   = lane & 15;
    const float b1v = b1[0];

    // Load all 8 B fragments once (contiguous 32 B per lane each)
    HF Bf[2][4];
    #pragma unroll
    for (int kt = 0; kt < 2; ++kt) {
        #pragma unroll
        for (int nt = 0; nt < 4; ++nt) {
            const _Float16* src = &sW0f[((kt * 4 + nt) * 32 + lane) * 16];
            #pragma unroll
            for (int i = 0; i < 16; ++i) Bf[kt][nt].e[i] = src[i];
        }
    }

    #pragma unroll
    for (int mt = 0; mt < 2; ++mt) {
        const int mbase = wave * 32 + mt * 16;  // point base within block
        HF A0, A1;
        #pragma unroll
        for (int i = 0; i < 16; ++i) {
            const int k = kmap(i, g);
            A0.e[i] = sFeats[(mbase + n0) * 64 + k];
            A1.e[i] = sFeats[(mbase + n0) * 64 + 32 + k];
        }
        float part[8];
        #pragma unroll
        for (int r = 0; r < 8; ++r) part[r] = 0.0f;

        #pragma unroll
        for (int nt = 0; nt < 4; ++nt) {
            v8f c = {};
            c = __builtin_amdgcn_wmma_f32_16x16x32_f16(
                    false, A0.v, false, Bf[0][nt].v, (short)0, c, false, false);
            c = __builtin_amdgcn_wmma_f32_16x16x32_f16(
                    false, A1.v, false, Bf[1][nt].v, (short)0, c, false, false);
            const int   n  = nt * 16 + n0;
            const float bb = b0[n];
            const float wv = W1[n];
            #pragma unroll
            for (int r = 0; r < 8; ++r) {
                const float h = softplus10(c[r] + bb);
                part[r] += h * wv;
            }
        }

        // Reduce across the 16 lanes of each half-wave; lanes n0==0 store.
        #pragma unroll
        for (int r = 0; r < 8; ++r) {
            float v = part[r];
            v += __shfl_xor(v, 1, 32);
            v += __shfl_xor(v, 2, 32);
            v += __shfl_xor(v, 4, 32);
            v += __shfl_xor(v, 8, 32);
            if (n0 == 0) {
                const int pp = blockIdx.x * PTS_PER_BLOCK + mbase + r + 8 * g;
                if (pp < P) {
                    const int tt = pp % T_SAMPLES;
                    const float delta = (tt == 0)
                                            ? (1.0f / (float)T_SAMPLES)
                                            : (2.0f / (float)(T_SAMPLES - 1));
                    out[pp] = delta * softplus10(v + b1v);
                }
            }
        }
    }
}

extern "C" void kernel_launch(void* const* d_in, const int* in_sizes, int n_in,
                              void* d_out, int out_size, void* d_ws,
                              size_t ws_size, hipStream_t stream) {
    const float* x   = (const float*)d_in[0];
    const float* emb = (const float*)d_in[1];
    const float* W0  = (const float*)d_in[2];
    const float* b0  = (const float*)d_in[3];
    const float* W1  = (const float*)d_in[4];
    const float* b1  = (const float*)d_in[5];
    float* out = (float*)d_out;
    const int P = out_size;  // B*N*T sample points
    const int blocks = (P + PTS_PER_BLOCK - 1) / PTS_PER_BLOCK;
    nerf_density_fused<<<blocks, PTS_PER_BLOCK, 0, stream>>>(
        x, emb, W0, b0, W1, b1, out, P);
}
